// Tacotron_19009525252566
// MI455X (gfx1250) — matrix operand order repaired and checked by reading, not compiled, not run on hardware
//
#include <hip/hip_runtime.h>
#include <stdint.h>

// ---------------- constants ----------------
#define BATCH 16
#define TC    200
#define TM    600
#define STEPS 300
#define EMB   512
#define ENCD  256
#define DECD  128
#define NMELS 80
#define FFTB  513
#define POSTD 512
#define LSTMD 1024
#define SPK   256
#define MAXR  20
#define RR    2
#define AF    32
#define AK    31
#define BN_SCALE 0.99999500003750f   /* 1/sqrt(1+1e-5) */

// ---------------- types for WMMA ----------------
typedef __attribute__((ext_vector_type(16))) __bf16 bf16x16;
typedef __attribute__((ext_vector_type(8)))  float  f32x8;

union BF16Frag { unsigned short u[16]; bf16x16 v; };
union F32Acc   { float f[8];           f32x8   v; };

__device__ __forceinline__ unsigned short f32_to_bf16(float f) {
  unsigned int u = __float_as_uint(f);
  u += 0x7FFFu + ((u >> 16) & 1u);           // round-to-nearest-even
  return (unsigned short)(u >> 16);
}
__device__ __forceinline__ unsigned int pack_bf16x2(float a, float b) {
  return (unsigned int)f32_to_bf16(a) | ((unsigned int)f32_to_bf16(b) << 16);
}
__device__ __forceinline__ float sigmoidf(float x) { return 1.f / (1.f + __expf(-x)); }
__device__ __forceinline__ unsigned int hash_u32(unsigned int x) {
  x ^= x >> 16; x *= 0x7feb352du; x ^= x >> 15; x *= 0x846ca68bu; x ^= x >> 16; return x;
}

// -------- WMMA GEMM: out[M,N] = act(A[M,K] @ W[N,K]^T + bias) (+= if acc) --------
// One wave computes a 16(M) x 64(N) output slab: A fragment reused across 4 WMMA tiles.
__global__ __launch_bounds__(32)
void k_linear_wmma(const float* __restrict__ A, const float* __restrict__ W,
                   const float* __restrict__ bias, float* __restrict__ out,
                   int M, int N, int K, int act, int acc) {
  __shared__ unsigned short As[16][34];   // [m][k], K-chunk of 32
  __shared__ unsigned short Bs[64][34];   // [n][k]
  const int lane  = threadIdx.x;          // 0..31, wave32
  const int ntile = blockIdx.x * 64;
  const int mtile = blockIdx.y * 16;
  const int col   = lane & 15;
  const int hi    = lane >> 4;

  F32Acc cacc[4];
  #pragma unroll
  for (int j = 0; j < 4; ++j)
    #pragma unroll
    for (int v = 0; v < 8; ++v) cacc[j].f[v] = 0.f;
  if (acc) {
    #pragma unroll
    for (int j = 0; j < 4; ++j)
      #pragma unroll
      for (int v = 0; v < 8; ++v) {
        int m = mtile + v + 8 * hi, n = ntile + 16 * j + col;
        if (m < M && n < N) cacc[j].f[v] = out[(size_t)m * N + n];
      }
  }

  const bool mn_full = (mtile + 16 <= M) && (ntile + 64 <= N);

  for (int k0 = 0; k0 < K; k0 += 32) {
    if (k0 + 32 < K) {  // per-lane spread prefetch of next K-slab -> global_prefetch_b8
      __builtin_prefetch(&A[(size_t)(mtile + col) * K + k0 + 32], 0, 1);
      __builtin_prefetch(&W[(size_t)(ntile + lane) * K + k0 + 32], 0, 1);
      __builtin_prefetch(&W[(size_t)(ntile + 32 + lane) * K + k0 + 32], 0, 1);
    }
    if (mn_full && (k0 + 32 <= K)) {
      // ---- fast staging: float4 loads, packed bf16x2 stores, no guards ----
      {
        int r = lane >> 1, kb = (lane & 1) * 16;
        const float4* src = (const float4*)(A + (size_t)(mtile + r) * K + k0 + kb);
        unsigned int* dst = (unsigned int*)&As[r][kb];
        #pragma unroll
        for (int q = 0; q < 4; ++q) {
          float4 f = src[q];
          dst[2 * q]     = pack_bf16x2(f.x, f.y);
          dst[2 * q + 1] = pack_bf16x2(f.z, f.w);
        }
      }
      #pragma unroll
      for (int rr = 0; rr < 2; ++rr) {
        int nn = 2 * lane + rr;
        const float4* src = (const float4*)(W + (size_t)(ntile + nn) * K + k0);
        unsigned int* dst = (unsigned int*)&Bs[nn][0];
        #pragma unroll
        for (int q = 0; q < 8; ++q) {
          float4 f = src[q];
          dst[2 * q]     = pack_bf16x2(f.x, f.y);
          dst[2 * q + 1] = pack_bf16x2(f.z, f.w);
        }
      }
    } else {
      // ---- guarded edge-tile staging ----
      for (int i = lane; i < 16 * 32; i += 32) {
        int r = i >> 5, kk = i & 31;
        int m = mtile + r, k = k0 + kk;
        As[r][kk] = f32_to_bf16((m < M && k < K) ? A[(size_t)m * K + k] : 0.f);
      }
      for (int i = lane; i < 64 * 32; i += 32) {
        int nn = i >> 5, kk = i & 31;
        int n = ntile + nn, k = k0 + kk;
        Bs[nn][kk] = f32_to_bf16((n < N && k < K) ? W[(size_t)n * K + k] : 0.f);
      }
    }
    __syncthreads();
    BF16Frag a;
    #pragma unroll
    for (int v = 0; v < 8; ++v) {
      int ka = (v < 4 ? 2 * v : 2 * v + 8) + 8 * hi;   // A 16x32 bf16 layout (ISA 7.12.2)
      a.u[2 * v]     = As[col][ka];
      a.u[2 * v + 1] = As[col][ka + 1];
    }
    #pragma unroll
    for (int j = 0; j < 4; ++j) {
      BF16Frag b;
      #pragma unroll
      for (int v = 0; v < 8; ++v) {
        int kb = (v < 4 ? 2 * v : 2 * v + 8) + 8 * hi; // B 32x16 bf16 layout
        b.u[2 * v]     = Bs[16 * j + col][kb];
        b.u[2 * v + 1] = Bs[16 * j + col][kb + 1];
      }
      cacc[j].v = __builtin_amdgcn_wmma_f32_16x16x32_bf16(false, a.v, false, b.v,
                                                          (short)0, cacc[j].v, false, false);
    }
    __syncthreads();
  }

  #pragma unroll
  for (int j = 0; j < 4; ++j)
    #pragma unroll
    for (int v = 0; v < 8; ++v) {
      int m = mtile + v + 8 * hi, n = ntile + 16 * j + col;
      if (m < M && n < N) {
        float r = cacc[j].f[v];
        if (bias) r += bias[n];
        if (act == 1) r = fmaxf(r, 0.f);
        else if (act == 2) r = sigmoidf(r);
        else if (act == 3) r = tanhf(r);
        out[(size_t)m * N + n] = r;
      }
    }
}

// ---------------- elementwise / data movement kernels ----------------
__global__ void k_zero(float* p, int n) {
  int i = blockIdx.x * blockDim.x + threadIdx.x; if (i < n) p[i] = 0.f;
}
__global__ void k_add(float* out, const float* a, const float* b, int n) {
  int i = blockIdx.x * blockDim.x + threadIdx.x; if (i < n) out[i] = a[i] + b[i];
}
__global__ void k_dropout(float* x, int n, unsigned int seed) {
  int i = blockIdx.x * blockDim.x + threadIdx.x;
  if (i < n) { unsigned int r = hash_u32((unsigned)i * 0x9E3779B9u + seed);
               x[i] = (r & 1u) ? x[i] * 2.f : 0.f; }
}
__global__ void k_gather_embed(const int* chars, const float* table, float* out, int rows) {
  int i = blockIdx.x * blockDim.x + threadIdx.x;
  int total = rows * EMB; if (i >= total) return;
  int e = i % EMB, r = i / EMB;
  out[i] = table[(size_t)chars[r] * EMB + e];
}
// [B,C,T] -> [B,T,C]
__global__ void k_transpose_ct(const float* x, float* out, int Bn, int C, int T) {
  int i = blockIdx.x * blockDim.x + threadIdx.x;
  int total = Bn * C * T; if (i >= total) return;
  int cc = i % C, t = (i / C) % T, b = i / (C * T);
  out[i] = x[((size_t)b * C + cc) * T + t];
}
// [B,T,C] -> [B,C,T]
__global__ void k_transpose_tc(const float* x, float* out, int Bn, int C, int T) {
  int i = blockIdx.x * blockDim.x + threadIdx.x;
  int total = Bn * C * T; if (i >= total) return;
  int t = i % T, cc = (i / T) % C, b = i / (C * T);
  out[i] = x[((size_t)b * T + t) * C + cc];
}
__global__ void k_conv1d(const float* __restrict__ x, const float* __restrict__ w,
                         const float* __restrict__ cbias,
                         const float* __restrict__ bng, const float* __restrict__ bnb,
                         float* __restrict__ out, int Bn, int Cin, int Cout, int T,
                         int ks, int pad, int relu_f, int bn_f, int CoutTot, int coff) {
  int i = blockIdx.x * blockDim.x + threadIdx.x;
  int total = Bn * Cout * T; if (i >= total) return;
  int t = i % T, co = (i / T) % Cout, b = i / (T * Cout);
  float s = 0.f;
  const float* wp = w + (size_t)co * Cin * ks;
  for (int ci = 0; ci < Cin; ++ci) {
    const float* xp = x + ((size_t)b * Cin + ci) * T;
    for (int j = 0; j < ks; ++j) {
      int tt = t - pad + j;
      if (tt >= 0 && tt < T) s += xp[tt] * wp[ci * ks + j];
    }
  }
  if (cbias) s += cbias[co];
  if (relu_f) s = fmaxf(s, 0.f);
  if (bn_f)   s = s * (bng[co] * BN_SCALE) + bnb[co];
  out[((size_t)b * CoutTot + coff + co) * T + t] = s;
}
__global__ void k_maxpool2(const float* x, float* out, int n, int T) {
  int i = blockIdx.x * blockDim.x + threadIdx.x; if (i >= n) return;
  int t = i % T;
  float a = x[i], b = (t > 0) ? x[i - 1] : -1e30f;
  out[i] = fmaxf(a, b);
}
__global__ void k_highway(const float* x, const float* x1, const float* graw, float* out, int n) {
  int i = blockIdx.x * blockDim.x + threadIdx.x; if (i >= n) return;
  float g = sigmoidf(graw[i]);
  out[i] = g * fmaxf(x1[i], 0.f) + (1.f - g) * x[i];
}
// per-batch sequential GRU scan, h in LDS
__global__ void k_gru_scan(const float* __restrict__ gi, const float* __restrict__ whh,
                           const float* __restrict__ bhh, float* __restrict__ y,
                           int T, int H, int rev, int ystride, int yoff) {
  __shared__ float h[256];
  int b = blockIdx.x, j = threadIdx.x;
  h[j] = 0.f;
  __syncthreads();
  for (int s = 0; s < T; ++s) {
    int t = rev ? (T - 1 - s) : s;
    const float* gir = gi + ((size_t)b * T + t) * 3 * H;
    float hr = bhh[j], hz = bhh[H + j], hn = bhh[2 * H + j];
    float hold = h[j];
    for (int k = 0; k < H; ++k) {
      float hk = h[k];
      hr += whh[(size_t)j * H + k] * hk;
      hz += whh[(size_t)(H + j) * H + k] * hk;
      hn += whh[(size_t)(2 * H + j) * H + k] * hk;
    }
    float r = sigmoidf(gir[j] + hr);
    float z = sigmoidf(gir[H + j] + hz);
    float n = tanhf(gir[2 * H + j] + r * hn);
    float hnew = (1.f - z) * n + z * hold;
    __syncthreads();
    h[j] = hnew;
    __syncthreads();
    y[((size_t)b * T + t) * ystride + yoff + j] = hnew;
  }
}
__global__ void k_concat2(const float* a, int wa, const float* b, int wb, float* out, int rows) {
  int w = wa + wb;
  int i = blockIdx.x * blockDim.x + threadIdx.x;
  if (i >= rows * w) return;
  int cc = i % w, r = i / w;
  out[i] = (cc < wa) ? a[(size_t)r * wa + cc] : b[(size_t)r * wb + (cc - wa)];
}
// enc_seq[b,t,:] = [x[b,t,0:C], spk[b,0:S]]
__global__ void k_concat_spk(const float* x, const float* spk, float* out, int T, int C, int S) {
  int w = C + S;
  int i = blockIdx.x * blockDim.x + threadIdx.x;
  if (i >= BATCH * T * w) return;
  int cc = i % w, t = (i / w) % T, b = i / (w * T);
  out[i] = (cc < C) ? x[((size_t)b * T + t) * C + cc] : spk[(size_t)b * S + (cc - C)];
}
__global__ void k_gru_combine(const float* gi, const float* gh, float* h, int Bn, int H) {
  int i = blockIdx.x * blockDim.x + threadIdx.x;
  if (i >= Bn * H) return;
  int j = i % H, b = i / H;
  const float* gir = gi + (size_t)b * 3 * H;
  const float* ghr = gh + (size_t)b * 3 * H;
  float r = sigmoidf(gir[j] + ghr[j]);
  float z = sigmoidf(gir[H + j] + ghr[H + j]);
  float n = tanhf(gir[2 * H + j] + r * ghr[2 * H + j]);
  h[i] = (1.f - z) * n + z * h[i];
}
__global__ void k_lstm_update(const float* g, float* h, float* c, int Bn, int H) {
  int i = blockIdx.x * blockDim.x + threadIdx.x;
  if (i >= Bn * H) return;
  int j = i % H, b = i / H;
  const float* gr = g + (size_t)b * 4 * H;
  float ig = sigmoidf(gr[j]), fg = sigmoidf(gr[H + j]);
  float gg = tanhf(gr[2 * H + j]), og = sigmoidf(gr[3 * H + j]);
  float cn = fg * c[i] + ig * gg;
  c[i] = cn;
  h[i] = og * tanhf(cn);
}
__global__ void k_lsa_conv(const float* cum, const float* w, const float* bias, float* loc) {
  int i = blockIdx.x * blockDim.x + threadIdx.x;
  if (i >= BATCH * AF * TC) return;
  int t = i % TC, f = (i / TC) % AF, b = i / (TC * AF);
  float s = bias[f];
  const float* wp = w + f * AK;
  for (int j = 0; j < AK; ++j) {
    int tt = t - (AK - 1) / 2 + j;
    if (tt >= 0 && tt < TC) s += cum[b * TC + tt] * wp[j];
  }
  loc[i] = s;
}
__global__ void k_lsa_u(const float* pq, const float* encp, const float* ploc,
                        const float* v, const int* chars, float* u) {
  int i = blockIdx.x * blockDim.x + threadIdx.x;
  if (i >= BATCH * TC) return;
  int b = i / TC;
  float s = 0.f;
  const float* pqb = pq + (size_t)b * DECD;
  const float* ep  = encp + (size_t)i * DECD;
  const float* pl  = ploc + (size_t)i * DECD;
  for (int d = 0; d < DECD; ++d) s += tanhf(pqb[d] + ep[d] + pl[d]) * v[d];
  u[i] = (chars[i] != 0) ? s : 0.f;
}
__global__ void k_softmax_attn(const float* u, float* scores, float* cum, float* attn, int step) {
  __shared__ float red[256];
  int b = blockIdx.x, t = threadIdx.x;
  float vv = (t < TC) ? u[b * TC + t] : -1e30f;
  red[t] = vv; __syncthreads();
  for (int s = 128; s > 0; s >>= 1) { if (t < s) red[t] = fmaxf(red[t], red[t + s]); __syncthreads(); }
  float mx = red[0]; __syncthreads();
  float e = (t < TC) ? __expf(vv - mx) : 0.f;
  red[t] = e; __syncthreads();
  for (int s = 128; s > 0; s >>= 1) { if (t < s) red[t] += red[t + s]; __syncthreads(); }
  float inv = 1.f / red[0];
  if (t < TC) {
    float p = e * inv;
    scores[b * TC + t] = p;
    cum[b * TC + t] += p;
    attn[((size_t)b * STEPS + step) * TC + t] = p;
  }
}
__global__ void k_ctx(const float* scores, const float* enc_seq, float* ctx, int C) {
  int i = blockIdx.x * blockDim.x + threadIdx.x;
  if (i >= BATCH * C) return;
  int cc = i % C, b = i / C;
  float s = 0.f;
  for (int t = 0; t < TC; ++t)
    s += scores[b * TC + t] * enc_seq[((size_t)b * TC + t) * C + cc];
  ctx[i] = s;
}
__global__ void k_pin(const float* mels, float* pin, int step) {
  int i = blockIdx.x * blockDim.x + threadIdx.x;
  if (i >= BATCH * NMELS) return;
  if (step == 0) { pin[i] = 0.f; return; }
  int m = i % NMELS, b = i / NMELS;
  pin[i] = mels[((size_t)b * NMELS + m) * TM + (step * RR - 1)];
}
__global__ void k_mel_out(const float* melfull, float* mel, int step) {
  int i = blockIdx.x * blockDim.x + threadIdx.x;
  if (i >= BATCH * NMELS * RR) return;
  int j = i % RR, m = (i / RR) % NMELS, b = i / (RR * NMELS);
  mel[((size_t)b * NMELS + m) * TM + step * RR + j] = melfull[(size_t)b * (NMELS * MAXR) + m * MAXR + j];
}
__global__ void k_stop(const float* x, const float* ctx, const float* w, const float* bias,
                       float* stop_out, int step) {
  int b = blockIdx.x * blockDim.x + threadIdx.x;
  if (b >= BATCH) return;
  float s = bias[0];
  for (int k = 0; k < LSTMD; ++k) s += x[(size_t)b * LSTMD + k] * w[k];
  for (int k = 0; k < ENCD + SPK; ++k) s += ctx[(size_t)b * (ENCD + SPK) + k] * w[LSTMD + k];
  s = sigmoidf(s);
  stop_out[(size_t)b * TM + step * RR]     = s;
  stop_out[(size_t)b * TM + step * RR + 1] = s;
}

// ---------------- host side ----------------
struct RNNP  { const float *bhh, *bih, *whh, *wih; };
struct ConvP { const float *b, *g, *w; };
struct HWP   { const float *b1, *b2, *w1, *w2; };
struct CBHGP { ConvP bank[5]; RNNP gru_b, gru_f; HWP hw[4]; ConvP p1, p2; const float* pre_hw; };

static inline dim3 nb(size_t n) { return dim3((unsigned)((n + 255) / 256)); }

extern "C" void kernel_launch(void* const* d_in, const int* in_sizes, int n_in,
                              void* d_out, int out_size, void* d_ws, size_t ws_size,
                              hipStream_t stream) {
  (void)in_sizes; (void)n_in; (void)out_size; (void)ws_size;
  // ---- unpack inputs (pytree sorted-key flatten order) ----
  int ix = 0;
  auto F = [&]() { return (const float*)d_in[ix++]; };
  const int*   chars = (const int*)d_in[ix++];
  const float* mels  = F();
  RNNP attn_rnn{F(), F(), F(), F()};
  const float *dec_fc1_b = F(), *dec_fc1_w = F(), *dec_fc2_b = F(), *dec_fc2_w = F();
  const float *embedding = F();
  CBHGP enc_cbhg; CBHGP post_cbhg;
  for (int k = 0; k < 5; ++k) { enc_cbhg.bank[k].b = F(); enc_cbhg.bank[k].g = F(); enc_cbhg.bank[k].w = F(); }
  enc_cbhg.gru_b = RNNP{F(), F(), F(), F()};
  enc_cbhg.gru_f = RNNP{F(), F(), F(), F()};
  for (int h = 0; h < 4; ++h) enc_cbhg.hw[h] = HWP{F(), F(), F(), F()};
  enc_cbhg.p1 = ConvP{F(), F(), F()};
  enc_cbhg.p2 = ConvP{F(), F(), F()};
  enc_cbhg.pre_hw = nullptr;
  const float *enc_fc1_b = F(), *enc_fc1_w = F(), *enc_fc2_b = F(), *enc_fc2_w = F();
  const float *encoder_proj_w = F();
  const float *lsa_L_w = F(), *lsa_W_b = F(), *lsa_W_w = F(),
              *lsa_conv_b = F(), *lsa_conv_w = F(), *lsa_v_w = F();
  RNNP lstm1{F(), F(), F(), F()}, lstm2{F(), F(), F(), F()};
  const float *mel_proj_w = F();
  for (int k = 0; k < 5; ++k) { post_cbhg.bank[k].b = F(); post_cbhg.bank[k].g = F(); post_cbhg.bank[k].w = F(); }
  post_cbhg.gru_b = RNNP{F(), F(), F(), F()};
  post_cbhg.gru_f = RNNP{F(), F(), F(), F()};
  for (int h = 0; h < 4; ++h) post_cbhg.hw[h] = HWP{F(), F(), F(), F()};
  post_cbhg.p1 = ConvP{F(), F(), F()};
  post_cbhg.p2 = ConvP{F(), F(), F()};
  post_cbhg.pre_hw = F();
  const float *post_proj_w = F();
  const float *rnn_in_b = F(), *rnn_in_w = F();
  const float *stop_b = F(), *stop_w = F();
  const float *speaker = F();

  // ---- output regions ----
  float* out_mel    = (float*)d_out;                          // [16,80,600]
  float* out_linear = out_mel + (size_t)BATCH * NMELS * TM;   // [16,513,600]
  float* out_attn   = out_linear + (size_t)BATCH * FFTB * TM; // [16,300,200]
  float* out_stop   = out_attn + (size_t)BATCH * STEPS * TC;  // [16,600]

  // ---- scratch bump allocator (256B aligned blocks) ----
  size_t off = 0;
  auto alloc = [&](size_t nfloats) -> float* {
    float* p = (float*)((char*)d_ws + off);
    off += ((nfloats * sizeof(float)) + 255) & ~(size_t)255;
    return p;
  };
  auto gemm = [&](const float* A, const float* W, const float* bias, float* out,
                  int M, int N, int K, int act, int acc) {
    dim3 grid((N + 63) / 64, (M + 15) / 16);
    k_linear_wmma<<<grid, 32, 0, stream>>>(A, W, bias, out, M, N, K, act, acc);
  };
  auto conv = [&](const float* x, const ConvP& c, const float* cbias, float* out,
                  int Cin, int Cout, int T, int ks, int pad, int relu_f, int bn_f,
                  int CoutTot, int coff) {
    size_t n = (size_t)BATCH * Cout * T;
    k_conv1d<<<nb(n), 256, 0, stream>>>(x, c.w, cbias, c.g, c.b, out, BATCH, Cin, Cout, T,
                                        ks, pad, relu_f, bn_f, CoutTot, coff);
  };

  // ---- CBHG (shared between encoder and postnet) ----
  auto run_cbhg = [&](const CBHGP& p, const float* x_bct, int Cin, int T, int ch,
                      int proj1) -> float* {
    const int K = 5, H2 = ch / 2;
    size_t n;
    float* bank = alloc((size_t)BATCH * ch * K * T);
    for (int k = 1; k <= K; ++k)
      conv(x_bct, p.bank[k - 1], nullptr, bank, Cin, ch, T, k, k / 2, 1, 1, ch * K, (k - 1) * ch);
    float* mp = alloc((size_t)BATCH * ch * K * T);
    n = (size_t)BATCH * ch * K * T;
    k_maxpool2<<<nb(n), 256, 0, stream>>>(bank, mp, (int)n, T);
    float* p1b = alloc((size_t)BATCH * ch * T);
    conv(mp, p.p1, nullptr, p1b, ch * K, ch, T, 3, 1, 1, 1, ch, 0);
    float* p2b = alloc((size_t)BATCH * proj1 * T);
    conv(p1b, p.p2, nullptr, p2b, ch, proj1, T, 3, 1, 0, 1, proj1, 0);
    n = (size_t)BATCH * proj1 * T;
    k_add<<<nb(n), 256, 0, stream>>>(p2b, p2b, x_bct, (int)n);
    float* px = alloc((size_t)BATCH * T * proj1);
    k_transpose_ct<<<nb(n), 256, 0, stream>>>(p2b, px, BATCH, proj1, T);
    int rows = BATCH * T;
    float* hx = px;
    if (p.pre_hw) { hx = alloc((size_t)rows * ch); gemm(px, p.pre_hw, nullptr, hx, rows, ch, proj1, 0, 0); }
    float* hx1 = alloc((size_t)rows * ch);
    float* hg  = alloc((size_t)rows * ch);
    float* hx2 = alloc((size_t)rows * ch);
    for (int h = 0; h < 4; ++h) {
      gemm(hx, p.hw[h].w1, p.hw[h].b1, hx1, rows, ch, ch, 0, 0);
      gemm(hx, p.hw[h].w2, p.hw[h].b2, hg,  rows, ch, ch, 0, 0);
      n = (size_t)rows * ch;
      k_highway<<<nb(n), 256, 0, stream>>>(hx, hx1, hg, hx2, (int)n);
      float* t2 = hx; hx = hx2; hx2 = t2;
    }
    float* gif = alloc((size_t)rows * 3 * H2);
    float* gib = alloc((size_t)rows * 3 * H2);
    gemm(hx, p.gru_f.wih, p.gru_f.bih, gif, rows, 3 * H2, ch, 0, 0);
    gemm(hx, p.gru_b.wih, p.gru_b.bih, gib, rows, 3 * H2, ch, 0, 0);
    float* y = alloc((size_t)rows * ch);
    k_gru_scan<<<BATCH, H2, 0, stream>>>(gif, p.gru_f.whh, p.gru_f.bhh, y, T, H2, 0, ch, 0);
    k_gru_scan<<<BATCH, H2, 0, stream>>>(gib, p.gru_b.whh, p.gru_b.bhh, y, T, H2, 1, ch, H2);
    return y;
  };

  // ================= ENCODER =================
  const int ROWS_E = BATCH * TC;
  float* emb = alloc((size_t)ROWS_E * EMB);
  k_gather_embed<<<nb((size_t)ROWS_E * EMB), 256, 0, stream>>>(chars, embedding, emb, ROWS_E);
  float* pre1 = alloc((size_t)ROWS_E * ENCD);
  float* pre2 = alloc((size_t)ROWS_E * ENCD);
  gemm(emb, enc_fc1_w, enc_fc1_b, pre1, ROWS_E, ENCD, EMB, 1, 0);
  k_dropout<<<nb((size_t)ROWS_E * ENCD), 256, 0, stream>>>(pre1, ROWS_E * ENCD, 0xE1u);
  gemm(pre1, enc_fc2_w, enc_fc2_b, pre2, ROWS_E, ENCD, ENCD, 1, 0);
  k_dropout<<<nb((size_t)ROWS_E * ENCD), 256, 0, stream>>>(pre2, ROWS_E * ENCD, 0xE2u);
  float* xct = alloc((size_t)BATCH * ENCD * TC);
  k_transpose_tc<<<nb((size_t)BATCH * ENCD * TC), 256, 0, stream>>>(pre2, xct, BATCH, ENCD, TC);
  float* cb_out = run_cbhg(enc_cbhg, xct, ENCD, TC, ENCD, ENCD);   // [B*Tc, 256]
  float* enc_seq  = alloc((size_t)ROWS_E * (ENCD + SPK));
  k_concat_spk<<<nb((size_t)ROWS_E * (ENCD + SPK)), 256, 0, stream>>>(cb_out, speaker, enc_seq, TC, ENCD, SPK);
  float* enc_proj = alloc((size_t)ROWS_E * DECD);
  gemm(enc_seq, encoder_proj_w, nullptr, enc_proj, ROWS_E, DECD, ENCD + SPK, 0, 0);

  // ================= DECODER =================
  const int PD = 2 * DECD;                 // 256
  const int CTXD = ENCD + SPK;             // 512
  float* ah = alloc(BATCH * DECD);
  float* h1 = alloc(BATCH * LSTMD); float* c1 = alloc(BATCH * LSTMD);
  float* h2 = alloc(BATCH * LSTMD); float* c2 = alloc(BATCH * LSTMD);
  float* ctx = alloc(BATCH * CTXD);
  float* cum = alloc(BATCH * TC);
  k_zero<<<nb(BATCH * DECD), 256, 0, stream>>>(ah, BATCH * DECD);
  k_zero<<<nb(BATCH * LSTMD), 256, 0, stream>>>(h1, BATCH * LSTMD);
  k_zero<<<nb(BATCH * LSTMD), 256, 0, stream>>>(c1, BATCH * LSTMD);
  k_zero<<<nb(BATCH * LSTMD), 256, 0, stream>>>(h2, BATCH * LSTMD);
  k_zero<<<nb(BATCH * LSTMD), 256, 0, stream>>>(c2, BATCH * LSTMD);
  k_zero<<<nb(BATCH * CTXD), 256, 0, stream>>>(ctx, BATCH * CTXD);
  k_zero<<<nb(BATCH * TC), 256, 0, stream>>>(cum, BATCH * TC);

  float* pin  = alloc(BATCH * NMELS);
  float* po1  = alloc(BATCH * PD);
  float* po2  = alloc(BATCH * PD);
  float* catA = alloc(BATCH * (CTXD + PD));
  float* giA  = alloc(BATCH * 3 * DECD);
  float* ghA  = alloc(BATCH * 3 * DECD);
  float* pq   = alloc(BATCH * DECD);
  float* loc  = alloc((size_t)BATCH * AF * TC);
  float* locT = alloc((size_t)BATCH * TC * AF);
  float* ploc = alloc((size_t)BATCH * TC * DECD);
  float* u    = alloc(BATCH * TC);
  float* scores = alloc(BATCH * TC);
  float* catR = alloc(BATCH * (CTXD + DECD));
  float* xr   = alloc(BATCH * LSTMD);
  float* xr2  = alloc(BATCH * LSTMD);
  float* xr3  = alloc(BATCH * LSTMD);
  float* gl   = alloc(BATCH * 4 * LSTMD);
  float* melfull = alloc(BATCH * NMELS * MAXR);

  for (int step = 0; step < STEPS; ++step) {
    // teacher-forced prenet input
    k_pin<<<nb(BATCH * NMELS), 256, 0, stream>>>(mels, pin, step);
    gemm(pin, dec_fc1_w, dec_fc1_b, po1, BATCH, PD, NMELS, 1, 0);
    k_dropout<<<nb(BATCH * PD), 256, 0, stream>>>(po1, BATCH * PD, 0xD000u + 2u * step);
    gemm(po1, dec_fc2_w, dec_fc2_b, po2, BATCH, PD, PD, 1, 0);
    k_dropout<<<nb(BATCH * PD), 256, 0, stream>>>(po2, BATCH * PD, 0xD001u + 2u * step);
    // attention GRU
    k_concat2<<<nb(BATCH * (CTXD + PD)), 256, 0, stream>>>(ctx, CTXD, po2, PD, catA, BATCH);
    gemm(catA, attn_rnn.wih, attn_rnn.bih, giA, BATCH, 3 * DECD, CTXD + PD, 0, 0);
    gemm(ah,   attn_rnn.whh, attn_rnn.bhh, ghA, BATCH, 3 * DECD, DECD, 0, 0);
    k_gru_combine<<<nb(BATCH * DECD), 256, 0, stream>>>(giA, ghA, ah, BATCH, DECD);
    // location-sensitive attention
    gemm(ah, lsa_W_w, lsa_W_b, pq, BATCH, DECD, DECD, 0, 0);
    k_lsa_conv<<<nb((size_t)BATCH * AF * TC), 256, 0, stream>>>(cum, lsa_conv_w, lsa_conv_b, loc);
    k_transpose_ct<<<nb((size_t)BATCH * AF * TC), 256, 0, stream>>>(loc, locT, BATCH, AF, TC);
    gemm(locT, lsa_L_w, nullptr, ploc, BATCH * TC, DECD, AF, 0, 0);
    k_lsa_u<<<nb(BATCH * TC), 256, 0, stream>>>(pq, enc_proj, ploc, lsa_v_w, chars, u);
    k_softmax_attn<<<BATCH, 256, 0, stream>>>(u, scores, cum, out_attn, step);
    k_ctx<<<nb(BATCH * CTXD), 256, 0, stream>>>(scores, enc_seq, ctx, CTXD);
    // decoder RNN stack
    k_concat2<<<nb(BATCH * (CTXD + DECD)), 256, 0, stream>>>(ctx, CTXD, ah, DECD, catR, BATCH);
    gemm(catR, rnn_in_w, rnn_in_b, xr, BATCH, LSTMD, CTXD + DECD, 0, 0);
    gemm(xr, lstm1.wih, lstm1.bih, gl, BATCH, 4 * LSTMD, LSTMD, 0, 0);
    gemm(h1, lstm1.whh, lstm1.bhh, gl, BATCH, 4 * LSTMD, LSTMD, 0, 1);
    k_lstm_update<<<nb(BATCH * LSTMD), 256, 0, stream>>>(gl, h1, c1, BATCH, LSTMD);
    k_add<<<nb(BATCH * LSTMD), 256, 0, stream>>>(xr2, xr, h1, BATCH * LSTMD);
    gemm(xr2, lstm2.wih, lstm2.bih, gl, BATCH, 4 * LSTMD, LSTMD, 0, 0);
    gemm(h2, lstm2.whh, lstm2.bhh, gl, BATCH, 4 * LSTMD, LSTMD, 0, 1);
    k_lstm_update<<<nb(BATCH * LSTMD), 256, 0, stream>>>(gl, h2, c2, BATCH, LSTMD);
    k_add<<<nb(BATCH * LSTMD), 256, 0, stream>>>(xr3, xr2, h2, BATCH * LSTMD);
    // outputs
    gemm(xr3, mel_proj_w, nullptr, melfull, BATCH, NMELS * MAXR, LSTMD, 0, 0);
    k_mel_out<<<nb(BATCH * NMELS * RR), 256, 0, stream>>>(melfull, out_mel, step);
    k_stop<<<1, BATCH, 0, stream>>>(xr3, ctx, stop_w, stop_b, out_stop, step);
  }

  // ================= POSTNET =================
  float* post_out = run_cbhg(post_cbhg, out_mel, NMELS, TM, POSTD, NMELS);  // [B*Tm, 512]
  float* postlin = alloc((size_t)BATCH * TM * FFTB);
  gemm(post_out, post_proj_w, nullptr, postlin, BATCH * TM, FFTB, POSTD, 0, 0);
  k_transpose_tc<<<nb((size_t)BATCH * FFTB * TM), 256, 0, stream>>>(postlin, out_linear, BATCH, FFTB, TM);
}
